// RecurrentDecoder_33139967656648
// MI455X (gfx1250) — matrix operand order repaired and checked by reading, compiled, not run on hardware
//
#include <hip/hip_runtime.h>
#include <hip/hip_bf16.h>
#include <math.h>

// ---------------------------------------------------------------------------
// CDNA5 (gfx1250) WMMA types
// ---------------------------------------------------------------------------
typedef __attribute__((ext_vector_type(16))) _Float16 v16h;
typedef __attribute__((ext_vector_type(8)))  float    v8f;

#define WMMA_F16(a, b, c) \
  __builtin_amdgcn_wmma_f32_16x16x32_f16(false, (a), false, (b), (short)0, (c), false, false)

// Problem constants
#define TT 32
#define BB 32
#define SS 64
#define HH 128
#define VV 32000
#define G4H 512   // 4*H

__device__ __forceinline__ float sigmoidf_(float x) {
  return 1.0f / (1.0f + __expf(-x));
}

// ---------------------------------------------------------------------------
// Fragment loaders (contiguous k order: element s of lane half h is k = 16h + s, wave32)
// ---------------------------------------------------------------------------
__device__ __forceinline__ v16h load_a_from(const _Float16* arow, int k0) {
  union { v16h v; uint4 q[2]; } a;
  a.q[0] = *(const uint4*)(arow + k0);
  a.q[1] = *(const uint4*)(arow + k0 + 8);
  return a.v;
}

__device__ __forceinline__ v16h load_b_from(const _Float16* brow, int k0) {
  union { v16h v; uint4 q[2]; } b;
  b.q[0] = *(const uint4*)(brow + k0);
  b.q[1] = *(const uint4*)(brow + k0 + 8);
  return b.v;
}

__device__ __forceinline__ void store_d_lds_f16(_Float16* D, int ldd, int m_t, int n0, v8f acc) {
  const int lane = threadIdx.x & 31;
  const int g = lane >> 4, n = lane & 15;
#pragma unroll
  for (int r = 0; r < 8; ++r)
    D[(size_t)((m_t << 4) + (g << 3) + r) * ldd + n0 + n] = (_Float16)acc[r];
}

// gates[32,512] = cat(x,h)[32,Ktot] @ Bmat, Bmat[k][n] = (k<in_dim? wih[n][k] : whh[n][k-in_dim])
__device__ __forceinline__ void gemm_gates(const _Float16* xbuf, int ldx,
                                           const _Float16* wih, int in_dim,
                                           const _Float16* whh,
                                           _Float16* gout, int ldg) {
  const int wave = threadIdx.x >> 5;
  const int lane = threadIdx.x & 31;
  const int g = lane >> 4, l15 = lane & 15;
#pragma unroll 1
  for (int i = 0; i < 8; ++i) {
    const int flat = wave * 8 + i;           // 0..63 tiles (2 M x 32 N)
    const int m_t = flat & 1;
    const int n0 = (flat >> 1) << 4;
    const _Float16* arow = xbuf + (size_t)((m_t << 4) + l15) * ldx + (g << 4);
    const _Float16* bi   = wih  + (size_t)(n0 + l15) * in_dim + (g << 4);
    const _Float16* bh   = whh  + (size_t)(n0 + l15) * HH     + (g << 4);
    v8f acc = {};
#pragma unroll 2
    for (int k0 = 0; k0 < in_dim; k0 += 32)
      acc = WMMA_F16(load_a_from(arow, k0), load_b_from(bi, k0), acc);
#pragma unroll 2
    for (int k0 = 0; k0 < HH; k0 += 32)
      acc = WMMA_F16(load_a_from(arow, in_dim + k0), load_b_from(bh, k0), acc);
    store_d_lds_f16(gout, ldg, m_t, n0, acc);
  }
}

// ---------------------------------------------------------------------------
// Cross-block pipeline sync helpers (producer/consumer progress flags in ws)
// ---------------------------------------------------------------------------
__device__ __forceinline__ void pipe_wait(const int* flag, int target) {
  if (threadIdx.x == 0) {
    while (__hip_atomic_load(flag, __ATOMIC_ACQUIRE, __HIP_MEMORY_SCOPE_AGENT) < target)
      __builtin_amdgcn_s_sleep(1);
  }
  __syncthreads();
  __builtin_amdgcn_fence(__ATOMIC_ACQUIRE, "agent");
}

__device__ __forceinline__ void pipe_signal(int* flag, int value) {
  __threadfence();        // make this thread's global stores agent-visible
  __syncthreads();        // every thread has fenced its stores
  if (threadIdx.x == 0)
    __hip_atomic_store(flag, value, __ATOMIC_RELEASE, __HIP_MEMORY_SCOPE_AGENT);
}

// ---------------------------------------------------------------------------
// Small utility kernels
// ---------------------------------------------------------------------------
__global__ void __launch_bounds__(256) f32_to_f16_kernel(const float* __restrict__ src,
                                                         _Float16* __restrict__ dst, int n) {
  int i = blockIdx.x * 256 + threadIdx.x;
  if (i < n) dst[i] = (_Float16)src[i];
}

__global__ void __launch_bounds__(256) embed_kernel(const int* __restrict__ inputs,
                                                    const float* __restrict__ embed,
                                                    float* __restrict__ x0) {
  int i = blockIdx.x * 256 + threadIdx.x;   // T*B*H
  if (i >= TT * BB * HH) return;
  int tb = i >> 7, h = i & 127;
  x0[i] = embed[(size_t)inputs[tb] * HH + h];
}

__global__ void __launch_bounds__(64) zero_flags_kernel(int* flags) {
  if (threadIdx.x < 8) flags[threadIdx.x] = 0;
}

// ---------------------------------------------------------------------------
// Pipelined recurrent decoder: 8 blocks, block s = layer stage s.
// Stage 0: LSTM cell + attention -> dec.  Stages 1..7: lstms[0..6].
// Stage s waits for stage s-1's timestep t before consuming it.
// ---------------------------------------------------------------------------
struct PipeArgs {
  const float* x0;
  const float* enc;
  const _Float16* wih0; const _Float16* whh0;
  const float* bih0;    const float* bhh0;
  const _Float16* win16; const _Float16* wout16;
  const _Float16* lwih[7]; const _Float16* lwhh[7];
  const float*    lbih[7]; const float*    lbhh[7];
  float* dec;
  float* xs[7];     // stage s (s>=1) writes xs[s-1]
  int*   flags;     // [8] progress counters
};

__device__ void attn_body(const PipeArgs& a,
                          _Float16 (&xbuf)[BB][384],
                          _Float16 (&gbuf)[BB][G4H],
                          float* bsum) {
  const int tid = threadIdx.x;
  float c[16];
#pragma unroll
  for (int r = 0; r < 16; ++r) c[r] = 0.f;

  for (int i = tid; i < G4H; i += 256) bsum[i] = a.bih0[i] + a.bhh0[i];
  for (int i = tid; i < BB * HH; i += 256)
    xbuf[i >> 7][HH + (i & 127)] = (_Float16)0.f;   // h0 = 0

#pragma unroll 1
  for (int t = 0; t < TT; ++t) {
    // ---- stage x_t (f32 -> f16 in LDS)
#pragma unroll 1
    for (int i = tid; i < BB * HH; i += 256) {
      int b = i >> 7, h = i & 127;
      xbuf[b][h] = (_Float16)a.x0[((size_t)t * BB + b) * HH + h];
    }
    __syncthreads();

    gemm_gates(&xbuf[0][0], 384, a.wih0, HH, a.whh0, &gbuf[0][0], G4H);
    __syncthreads();

    // ---- LSTM activations (c in registers), h -> LDS f16
#pragma unroll
    for (int r = 0; r < 16; ++r) {
      int idx = tid + 256 * r;
      int b = idx >> 7, h = idx & 127;
      float ig = (float)gbuf[b][h]          + bsum[h];
      float fg = (float)gbuf[b][HH + h]     + bsum[HH + h];
      float gg = (float)gbuf[b][2 * HH + h] + bsum[2 * HH + h];
      float og = (float)gbuf[b][3 * HH + h] + bsum[3 * HH + h];
      c[r] = sigmoidf_(fg) * c[r] + sigmoidf_(ig) * tanhf(gg);
      float hv = sigmoidf_(og) * tanhf(c[r]);
      xbuf[b][HH + h] = (_Float16)hv;
    }
    __syncthreads();

    // ---- q = h @ win.T -> gbuf[:,0:128] f16
    {
      const int wave = tid >> 5;
      const int lane = tid & 31, g = lane >> 4, l15 = lane & 15;
#pragma unroll 1
      for (int i = 0; i < 2; ++i) {
        const int flat = wave * 2 + i;
        const int m_t = flat & 1, n0 = (flat >> 1) << 4;
        const _Float16* arow = &xbuf[0][HH] + (size_t)((m_t << 4) + l15) * 384 + (g << 4);
        const _Float16* brow = a.win16 + (size_t)(n0 + l15) * HH + (g << 4);
        v8f acc = {};
#pragma unroll 2
        for (int k0 = 0; k0 < HH; k0 += 32)
          acc = WMMA_F16(load_a_from(arow, k0), load_b_from(brow, k0), acc);
        store_d_lds_f16(&gbuf[0][0], G4H, m_t, n0, acc);
      }
    }
    __syncthreads();

    // ---- scores[b,s] = <ctx[b,s,:], q[b,:]>
    {
      const int b = tid >> 3;
      const int s8 = (tid & 7) * 8;
      float sc[8];
#pragma unroll
      for (int j = 0; j < 8; ++j) sc[j] = 0.f;
#pragma unroll 2
      for (int h = 0; h < HH; h += 4) {
        float q0 = (float)gbuf[b][h + 0];
        float q1 = (float)gbuf[b][h + 1];
        float q2 = (float)gbuf[b][h + 2];
        float q3 = (float)gbuf[b][h + 3];
#pragma unroll
        for (int j = 0; j < 8; ++j) {
          const float4 e = *(const float4*)&a.enc[((size_t)(s8 + j) * BB + b) * HH + h];
          sc[j] += q0 * e.x + q1 * e.y + q2 * e.z + q3 * e.w;
        }
      }
      float* srow = (float*)&xbuf[b][256];
#pragma unroll
      for (int j = 0; j < 8; ++j) srow[s8 + j] = sc[j];
    }
    __syncthreads();

    // ---- softmax (threads 0..31, one row each)
    if (tid < BB) {
      float* srow = (float*)&xbuf[tid][256];
      float mx = srow[0];
#pragma unroll 1
      for (int s = 1; s < SS; ++s) mx = fmaxf(mx, srow[s]);
      float sum = 0.f;
#pragma unroll 1
      for (int s = 0; s < SS; ++s) { float e = __expf(srow[s] - mx); srow[s] = e; sum += e; }
      float inv = 1.f / sum;
#pragma unroll 1
      for (int s = 0; s < SS; ++s) srow[s] *= inv;
    }
    __syncthreads();

    // ---- wc[b,h] = sum_s a[b,s]*ctx[b,s,h] -> xbuf[:,0:128]
#pragma unroll 1
    for (int r = 0; r < 16; ++r) {
      int idx = tid + 256 * r;
      int b = idx >> 7, h = idx & 127;
      const float* arow = (const float*)&xbuf[b][256];
      float acc = 0.f;
#pragma unroll 4
      for (int s = 0; s < SS; ++s)
        acc += arow[s] * a.enc[((size_t)s * BB + b) * HH + h];
      xbuf[b][h] = (_Float16)acc;
    }
    __syncthreads();

    // ---- out = tanh(cat(wc,h) @ wout.T) -> dec[t]
    {
      const int wave = tid >> 5;
      const int lane = tid & 31, g = lane >> 4, l15 = lane & 15;
#pragma unroll 1
      for (int i = 0; i < 2; ++i) {
        const int flat = wave * 2 + i;
        const int m_t = flat & 1, n0 = (flat >> 1) << 4;
        const _Float16* arow = &xbuf[0][0] + (size_t)((m_t << 4) + l15) * 384 + (g << 4);
        const _Float16* brow = a.wout16 + (size_t)(n0 + l15) * (2 * HH) + (g << 4);
        v8f acc = {};
#pragma unroll 2
        for (int k0 = 0; k0 < 2 * HH; k0 += 32)
          acc = WMMA_F16(load_a_from(arow, k0), load_b_from(brow, k0), acc);
#pragma unroll
        for (int r = 0; r < 8; ++r) {
          int m = (m_t << 4) + (g << 3) + r;
          a.dec[((size_t)t * BB + m) * HH + n0 + l15] = tanhf(acc[r]);
        }
      }
    }
    // dec[t] complete -> publish progress
    pipe_signal(&a.flags[0], t + 1);
  }
}

__device__ void lstm_body(const PipeArgs& a, int stage,
                          _Float16 (&xbuf)[BB][384],
                          _Float16 (&gbuf)[BB][G4H],
                          float* bsum) {
  const int tid = threadIdx.x;
  const int l = stage - 1;                 // lstms[l]
  const int use_dec = (stage >= 2);
  const int in_dim = use_dec ? 2 * HH : HH;
  const float* x_in = (stage == 1) ? a.dec : a.xs[stage - 2];
  const float* residual = x_in;            // valid only when use_dec
  float* x_out = a.xs[stage - 1];
  const _Float16* wih = a.lwih[l];
  const _Float16* whh = a.lwhh[l];
  const int* flag_prev = &a.flags[stage - 1];

  float c[16];
#pragma unroll
  for (int r = 0; r < 16; ++r) c[r] = 0.f;

  for (int i = tid; i < G4H; i += 256) bsum[i] = a.lbih[l][i] + a.lbhh[l][i];
  for (int i = tid; i < BB * HH; i += 256)
    xbuf[i >> 7][in_dim + (i & 127)] = (_Float16)0.f;   // h0 = 0

#pragma unroll 1
  for (int t = 0; t < TT; ++t) {
    pipe_wait(flag_prev, t + 1);           // upstream timestep t ready

    // stage cat(x, dec) as f16
#pragma unroll 1
    for (int i = tid; i < BB * in_dim; i += 256) {
      int b = use_dec ? (i >> 8) : (i >> 7);
      int k = use_dec ? (i & 255) : (i & 127);
      float v = (k < HH) ? x_in[((size_t)t * BB + b) * HH + k]
                         : a.dec[((size_t)t * BB + b) * HH + (k - HH)];
      xbuf[b][k] = (_Float16)v;
    }
    __syncthreads();

    gemm_gates(&xbuf[0][0], 384, wih, in_dim, whh, &gbuf[0][0], G4H);
    __syncthreads();

#pragma unroll
    for (int r = 0; r < 16; ++r) {
      int idx = tid + 256 * r;
      int b = idx >> 7, h = idx & 127;
      float ig = (float)gbuf[b][h]          + bsum[h];
      float fg = (float)gbuf[b][HH + h]     + bsum[HH + h];
      float gg = (float)gbuf[b][2 * HH + h] + bsum[2 * HH + h];
      float og = (float)gbuf[b][3 * HH + h] + bsum[3 * HH + h];
      c[r] = sigmoidf_(fg) * c[r] + sigmoidf_(ig) * tanhf(gg);
      float hv = sigmoidf_(og) * tanhf(c[r]);
      xbuf[b][in_dim + h] = (_Float16)hv;
      float o = hv + (use_dec ? residual[((size_t)t * BB + b) * HH + h] : 0.f);
      x_out[((size_t)t * BB + b) * HH + h] = o;
    }
    pipe_signal(&a.flags[stage], t + 1);
  }
}

__global__ void __launch_bounds__(256) pipeline_kernel(PipeArgs args) {
  __shared__ _Float16 xbuf[BB][384];
  __shared__ _Float16 gbuf[BB][G4H];
  __shared__ float    bsum[G4H];
  if (blockIdx.x == 0) attn_body(args, xbuf, gbuf, bsum);
  else                 lstm_body(args, blockIdx.x, xbuf, gbuf, bsum);
}

// ---------------------------------------------------------------------------
// Classifier: [1024,128] @ cls_w.T[128,32000] + b. Grid (32,125), 256 thr.
// ---------------------------------------------------------------------------
__global__ void __launch_bounds__(256) classifier_kernel(
    const float* __restrict__ x,        // [1024,128]
    const _Float16* __restrict__ w16,   // [32000,128] f16
    const float* __restrict__ bias,     // [32000]
    float* __restrict__ out)            // [1024,32000]
{
  __shared__ _Float16 xs[BB][HH];
  const int tid = threadIdx.x;
  const int mb = blockIdx.x;          // 0..31
  const int nb = blockIdx.y;          // 0..124

  for (int i = tid; i < BB * HH; i += 256) {
    int m = i >> 7, k = i & 127;
    xs[m][k] = (_Float16)x[((size_t)(mb * BB + m)) * HH + k];
  }
  __syncthreads();

  const int wave = tid >> 5, lane = tid & 31, g = lane >> 4, l15 = lane & 15;
  const int m_t = wave & 1;
  const int nbase = nb * 256 + (wave >> 1) * 64;

  // A fragments once (4 K-steps), reused by all 4 N-tiles
  const _Float16* arow = &xs[0][0] + (size_t)((m_t << 4) + l15) * HH + (g << 4);
  v16h afr[4];
#pragma unroll
  for (int k = 0; k < 4; ++k) afr[k] = load_a_from(arow, k * 32);

#pragma unroll 1
  for (int j = 0; j < 4; ++j) {
    const int n0 = nbase + j * 16;
    const _Float16* brow = w16 + (size_t)(n0 + l15) * HH + (g << 4);
    __builtin_prefetch(brow + HH * 16, 0, 1);   // next tile's row
    v8f acc = {};
#pragma unroll
    for (int k = 0; k < 4; ++k)
      acc = WMMA_F16(afr[k], load_b_from(brow, k * 32), acc);
    const float bv = bias[n0 + l15];
#pragma unroll
    for (int r = 0; r < 8; ++r) {
      int m = mb * BB + (m_t << 4) + (g << 3) + r;
      out[(size_t)m * VV + n0 + l15] = acc[r] + bv;
    }
  }
}

// ---------------------------------------------------------------------------
// Host-side orchestration
// ---------------------------------------------------------------------------
extern "C" void kernel_launch(void* const* d_in, const int* in_sizes, int n_in,
                              void* d_out, int out_size, void* d_ws, size_t ws_size,
                              hipStream_t stream) {
  (void)in_sizes; (void)n_in; (void)out_size; (void)ws_size;

  const int*   inputs = (const int*)d_in[0];
  const float* enc    = (const float*)d_in[1];
  const float* embed  = (const float*)d_in[2];
  const float* wih0   = (const float*)d_in[3];
  const float* whh0   = (const float*)d_in[4];
  const float* bih0   = (const float*)d_in[5];
  const float* bhh0   = (const float*)d_in[6];
  const float* win    = (const float*)d_in[7];
  const float* wout   = (const float*)d_in[8];
  const float* lw[7][4];
  for (int l = 0; l < 7; ++l)
    for (int k = 0; k < 4; ++k)
      lw[l][k] = (const float*)d_in[9 + 4 * l + k];
  const float* cls_w = (const float*)d_in[37];
  const float* cls_b = (const float*)d_in[38];
  float* out = (float*)d_out;

  // Workspace carve-up
  char* ws = (char*)d_ws;
  size_t off = 0;
  auto alloc = [&](size_t bytes) -> void* {
    void* p = ws + off;
    off += (bytes + 255) & ~(size_t)255;
    return p;
  };
  _Float16* wih0_16 = (_Float16*)alloc((size_t)G4H * HH * 2);
  _Float16* whh0_16 = (_Float16*)alloc((size_t)G4H * HH * 2);
  _Float16* win16   = (_Float16*)alloc((size_t)HH * HH * 2);
  _Float16* wout16  = (_Float16*)alloc((size_t)HH * 2 * HH * 2);
  _Float16* lwih16[7];
  _Float16* lwhh16[7];
  for (int l = 0; l < 7; ++l) {
    int in_dim = (l == 0) ? HH : 2 * HH;
    lwih16[l] = (_Float16*)alloc((size_t)G4H * in_dim * 2);
    lwhh16[l] = (_Float16*)alloc((size_t)G4H * HH * 2);
  }
  _Float16* clsw16 = (_Float16*)alloc((size_t)VV * HH * 2);
  float* x0  = (float*)alloc((size_t)TT * BB * HH * 4);
  float* dec = (float*)alloc((size_t)TT * BB * HH * 4);
  float* xs[7];
  for (int l = 0; l < 7; ++l) xs[l] = (float*)alloc((size_t)TT * BB * HH * 4);
  int* flags = (int*)alloc(8 * sizeof(int));

  auto conv = [&](const float* src, _Float16* dst, int n) {
    f32_to_f16_kernel<<<(n + 255) / 256, 256, 0, stream>>>(src, dst, n);
  };
  conv(wih0, wih0_16, G4H * HH);
  conv(whh0, whh0_16, G4H * HH);
  conv(win,  win16,   HH * HH);
  conv(wout, wout16,  HH * 2 * HH);
  for (int l = 0; l < 7; ++l) {
    int in_dim = (l == 0) ? HH : 2 * HH;
    conv(lw[l][0], lwih16[l], G4H * in_dim);
    conv(lw[l][1], lwhh16[l], G4H * HH);
  }
  conv(cls_w, clsw16, VV * HH);

  // Embedding gather + flag reset
  embed_kernel<<<(TT * BB * HH + 255) / 256, 256, 0, stream>>>(inputs, embed, x0);
  zero_flags_kernel<<<1, 64, 0, stream>>>(flags);

  // Pipelined recurrent decoder: 8 co-resident blocks, one per layer stage.
  PipeArgs pa;
  pa.x0 = x0; pa.enc = enc;
  pa.wih0 = wih0_16; pa.whh0 = whh0_16;
  pa.bih0 = bih0; pa.bhh0 = bhh0;
  pa.win16 = win16; pa.wout16 = wout16;
  for (int l = 0; l < 7; ++l) {
    pa.lwih[l] = lwih16[l]; pa.lwhh[l] = lwhh16[l];
    pa.lbih[l] = lw[l][2];  pa.lbhh[l] = lw[l][3];
  }
  pa.dec = dec;
  for (int l = 0; l < 7; ++l) pa.xs[l] = xs[l];
  pa.flags = flags;
  pipeline_kernel<<<8, 256, 0, stream>>>(pa);

  // Classifier: logits = xs[6] @ cls_w.T + cls_b
  dim3 grid(32, VV / 256);   // (32, 125)
  classifier_kernel<<<grid, 256, 0, stream>>>(xs[6], clsw16, cls_b, out);
}
